// RefNetwork_11441792876792
// MI455X (gfx1250) — matrix operand order repaired and checked by reading, not compiled, not run on hardware
//
#include <hip/hip_runtime.h>

typedef _Float16 half_t;
typedef __attribute__((ext_vector_type(16))) _Float16 v16h;
typedef __attribute__((ext_vector_type(8)))  float    v8f;

__device__ __forceinline__ int imin(int a, int b) { return a < b ? a : b; }
__device__ __forceinline__ int imax(int a, int b) { return a > b ? a : b; }

// ---------------------------------------------------------------------------
// Utility kernels
// ---------------------------------------------------------------------------
__global__ void zero_kernel(float* __restrict__ p, long n) {
    long i = (long)blockIdx.x * blockDim.x + threadIdx.x;
    if (i < n) p[i] = 0.f;
}

__global__ void scatter_kernel(const int* __restrict__ coord,
                               const float* __restrict__ cemb,
                               half_t* __restrict__ x0, float* __restrict__ mask,
                               int Nv, int H, int W) {
    int i = blockIdx.x * blockDim.x + threadIdx.x;
    if (i >= Nv) return;
    int z = coord[i * 4 + 1], y = coord[i * 4 + 2], x = coord[i * 4 + 3];
    long v = (long)(z * H + y) * W + x;
    mask[v] = 1.f;
    #pragma unroll
    for (int c = 0; c < 16; ++c) x0[v * 16 + c] = (half_t)cemb[i * 16 + c];
}

__global__ void maskpool_kernel(const float* __restrict__ in, float* __restrict__ out,
                                int Do, int Ho, int Wo) {
    int i = blockIdx.x * blockDim.x + threadIdx.x;
    int n = Do * Ho * Wo;
    if (i >= n) return;
    int z = i / (Ho * Wo); int r = i % (Ho * Wo); int y = r / Wo; int x = r % Wo;
    int Hi = Ho * 2, Wi = Wo * 2;
    float m = 0.f;
    #pragma unroll
    for (int dz = 0; dz < 2; ++dz)
        #pragma unroll
        for (int dy = 0; dy < 2; ++dy)
            #pragma unroll
            for (int dx = 0; dx < 2; ++dx)
                m = fmaxf(m, in[((long)((z * 2 + dz) * Hi + (y * 2 + dy))) * Wi + (x * 2 + dx)]);
    out[i] = m;
}

// Repack weights into WMMA A-fragment order: [tap][kBlock][coutTile][lane][16 halves]
// transposed==0: w is (ntaps, Cin, Cout)   (conv DHWIO taps flattened)
// transposed==1: w is (Cout, Cin) row-major (FC weights), ntaps must be 1
__global__ void packw_kernel(const float* __restrict__ w, half_t* __restrict__ wp,
                             int ntaps, int Cin, int Cout, int transposed) {
    int nkc = (Cin + 31) >> 5;
    int nct = Cout >> 4;
    int total = ntaps * nkc * nct * 32;
    int i = blockIdx.x * blockDim.x + threadIdx.x;
    if (i >= total) return;
    int lane = i & 31;
    int rest = i >> 5;
    int ct  = rest % nct; rest /= nct;
    int kcb = rest % nkc; rest /= nkc;
    int t   = rest;
    int n = lane & 15, hi = lane >> 4;
    int m = ct * 16 + n;
    int koff = hi * 16;
    half_t* dst = wp + (long)i * 16;
    #pragma unroll
    for (int h = 0; h < 16; ++h) {
        int k  = koff + h;
        int ci = kcb * 32 + k;
        float v = 0.f;
        if (ci < Cin)
            v = transposed ? w[(long)m * Cin + ci] : w[((long)t * Cin + ci) * Cout + m];
        dst[h] = (half_t)v;
    }
}

// Trilinear grid-sample: vol is (D,H,W,C) f16 channel-contiguous, grid is (npts,3) xyz
__global__ void gsample_kernel(const half_t* __restrict__ vol, int D, int H, int W, int C,
                               const float* __restrict__ grid, int npts,
                               half_t* __restrict__ out, int ostride, int ooff) {
    int p = blockIdx.x;
    if (p >= npts) return;
    float gx = grid[p * 3 + 0], gy = grid[p * 3 + 1], gz = grid[p * 3 + 2];
    float fx = (gx + 1.f) * 0.5f * (W - 1);
    float fy = (gy + 1.f) * 0.5f * (H - 1);
    float fz = (gz + 1.f) * 0.5f * (D - 1);
    float x0f = floorf(fx), y0f = floorf(fy), z0f = floorf(fz);
    float tx = fx - x0f, ty = fy - y0f, tz = fz - z0f;
    int x0 = (int)x0f, y0 = (int)y0f, z0 = (int)z0f;
    long base[8]; float wgt[8];
    int idx = 0;
    #pragma unroll
    for (int dz = 0; dz < 2; ++dz) {
        int zi = z0 + dz; float wz = dz ? tz : 1.f - tz;
        #pragma unroll
        for (int dy = 0; dy < 2; ++dy) {
            int yi = y0 + dy; float wy = dy ? ty : 1.f - ty;
            #pragma unroll
            for (int dx = 0; dx < 2; ++dx) {
                int xi = x0 + dx; float wx = dx ? tx : 1.f - tx;
                bool valid = zi >= 0 && zi < D && yi >= 0 && yi < H && xi >= 0 && xi < W;
                int zc = imin(imax(zi, 0), D - 1);
                int yc = imin(imax(yi, 0), H - 1);
                int xc = imin(imax(xi, 0), W - 1);
                base[idx] = ((long)(zc * H + yc) * W + xc) * C;
                wgt[idx]  = valid ? wz * wy * wx : 0.f;
                ++idx;
            }
        }
    }
    for (int c = threadIdx.x; c < C; c += blockDim.x) {
        float s = 0.f;
        #pragma unroll
        for (int k = 0; k < 8; ++k) s += wgt[k] * (float)vol[base[k] + c];
        out[(long)p * ostride + ooff + c] = (half_t)s;
    }
}

__global__ void latfill_kernel(const float* __restrict__ lemb, const int* __restrict__ iidx,
                               half_t* __restrict__ ft, int stride, int off, int C) {
    int p = blockIdx.x;
    const float* lat = lemb + (long)iidx[0] * C;
    for (int c = threadIdx.x; c < C; c += blockDim.x)
        ft[(long)p * stride + off + c] = (half_t)lat[c];
}

__global__ void catvl_kernel(const float* __restrict__ vd, const float* __restrict__ lp,
                             half_t* __restrict__ v2) {
    int p = blockIdx.x;
    for (int c = threadIdx.x; c < 27; c += blockDim.x)
        v2[(long)p * 352 + 256 + c] = (half_t)vd[(long)p * 27 + c];
    for (int c = threadIdx.x; c < 63; c += blockDim.x)
        v2[(long)p * 352 + 283 + c] = (half_t)lp[(long)p * 63 + c];
}

__global__ void alpha_kernel(const half_t* __restrict__ h, const float* __restrict__ w,
                             const float* __restrict__ b, float* __restrict__ out, int N) {
    int p = blockIdx.x * blockDim.x + threadIdx.x;
    if (p >= N) return;
    const half_t* r = h + (long)p * 256;
    float s = b[0];
    for (int i = 0; i < 256; ++i) s += w[i] * (float)r[i];
    out[(long)p * 4 + 3] = s;
}

__global__ void rgb_kernel(const half_t* __restrict__ h3, const float* __restrict__ w,
                           const float* __restrict__ b, float* __restrict__ out, int N) {
    int p = blockIdx.x * blockDim.x + threadIdx.x;
    if (p >= N) return;
    const half_t* r = h3 + (long)p * 128;
    #pragma unroll
    for (int k = 0; k < 3; ++k) {
        float s = b[k];
        for (int i = 0; i < 128; ++i) s += w[k * 128 + i] * (float)r[i];
        out[(long)p * 4 + k] = s;
    }
}

// ---------------------------------------------------------------------------
// WMMA implicit-GEMM 3x3x3 SAME conv. in: (D,H,W,Cin) f16; wp: packed A frags.
// out[vox,co] = relu(conv*sc+sh)*mask.  One wave = 16(cout)x16(voxel) tile.
// Out-of-bounds taps read a 64B zero page -> unconditional vector loads.
// ---------------------------------------------------------------------------
__global__ void __launch_bounds__(128)
conv3x3_wmma(const half_t* __restrict__ in, const v16h* __restrict__ wp,
             const float* __restrict__ mask, const float* __restrict__ scale,
             const float* __restrict__ shift, const half_t* __restrict__ zbh,
             half_t* __restrict__ out, int D, int H, int W, int Cin, int Cout) {
    const int lane = threadIdx.x & 31;
    const int wv   = threadIdx.x >> 5;
    const int nvox = D * H * W;
    const int tile = blockIdx.x * 4 + wv;
    if (tile * 16 >= nvox) return;                      // wave-uniform
    const int ctile = blockIdx.y;
    const int nkc   = (Cin + 31) >> 5;
    const int nct   = Cout >> 4;
    const int n    = lane & 15;
    const int hi   = lane >> 4;
    const int vox  = tile * 16 + n;
    const bool vok = vox < nvox;
    const int vz = vox / (H * W);
    const int vr = vox % (H * W);
    const int vy = vr / W;
    const int vx = vr % W;

    v8f acc = {};
    for (int t = 0; t < 27; ++t) {
        const int dz = t / 9 - 1, dy = (t / 3) % 3 - 1, dx = t % 3 - 1;
        const int zz = vz + dz, yy = vy + dy, xx = vx + dx;
        const bool inb = vok && zz >= 0 && zz < D && yy >= 0 && yy < H && xx >= 0 && xx < W;
        const long  sidx = ((long)(zz * H + yy) * W + xx) * Cin;
        for (int kcb = 0; kcb < nkc; ++kcb) {
            const int kb = kcb * 32 + hi * 16;
            const half_t* sp = (inb && kb < Cin) ? (in + sidx + kb) : zbh;
            v16h bfrag = *(const v16h*)sp;
            v16h afrag = wp[((long)(t * nkc + kcb) * nct + ctile) * 32 + lane];
            acc = __builtin_amdgcn_wmma_f32_16x16x32_f16(false, afrag, false, bfrag,
                                                         (short)0, acc, false, false);
        }
    }
    if (vok) {
        const float mval = mask[vox];
        half_t* dst = out + (long)vox * Cout;
        const int mrow0 = hi * 8;
        #pragma unroll
        for (int v = 0; v < 8; ++v) {
            const int co = ctile * 16 + mrow0 + v;
            float r = acc[v];
            if (scale) r = r * scale[co] + shift[co];
            dst[co] = (half_t)(fmaxf(r, 0.f) * mval);
        }
    }
}

// Stride-2 2x2x2 VALID down-conv + relu. in: (2Do,2Ho,2Wo,Cin) f16.
__global__ void __launch_bounds__(128)
convdown_wmma(const half_t* __restrict__ in, const v16h* __restrict__ wp,
              const half_t* __restrict__ zbh, half_t* __restrict__ out,
              int Do, int Ho, int Wo, int Cin, int Cout) {
    const int lane = threadIdx.x & 31;
    const int wv   = threadIdx.x >> 5;
    const int nvox = Do * Ho * Wo;
    const int tile = blockIdx.x * 4 + wv;
    if (tile * 16 >= nvox) return;
    const int ctile = blockIdx.y;
    const int nkc   = (Cin + 31) >> 5;
    const int nct   = Cout >> 4;
    const int n    = lane & 15;
    const int hi   = lane >> 4;
    const int vox  = tile * 16 + n;
    const bool vok = vox < nvox;
    const int oz = vox / (Ho * Wo);
    const int r  = vox % (Ho * Wo);
    const int oy = r / Wo;
    const int ox = r % Wo;
    const int Hi = Ho * 2, Wi = Wo * 2;

    v8f acc = {};
    for (int t = 0; t < 8; ++t) {
        const int dz = t >> 2, dy = (t >> 1) & 1, dx = t & 1;
        const long sidx = ((long)((oz * 2 + dz) * Hi + (oy * 2 + dy)) * Wi + (ox * 2 + dx)) * Cin;
        for (int kcb = 0; kcb < nkc; ++kcb) {
            const int kb = kcb * 32 + hi * 16;
            const half_t* sp = (vok && kb < Cin) ? (in + sidx + kb) : zbh;
            v16h bfrag = *(const v16h*)sp;
            v16h afrag = wp[((long)(t * nkc + kcb) * nct + ctile) * 32 + lane];
            acc = __builtin_amdgcn_wmma_f32_16x16x32_f16(false, afrag, false, bfrag,
                                                         (short)0, acc, false, false);
        }
    }
    if (vok) {
        half_t* dst = out + (long)vox * Cout;
        const int mrow0 = hi * 8;
        #pragma unroll
        for (int v = 0; v < 8; ++v)
            dst[ctile * 16 + mrow0 + v] = (half_t)fmaxf(acc[v], 0.f);
    }
}

// WMMA GEMM for the per-point MLP: out[p,co] = act(A[co,:] . B[p,:] + bias[co])
// Ap: packed fragments (nkc x M/16 x 32 x v16h); B: (N,Kpad) f16 rows (pad zeroed).
__global__ void __launch_bounds__(128)
gemm_wmma(const v16h* __restrict__ Ap, const half_t* __restrict__ B,
          const float* __restrict__ bias, half_t* __restrict__ out,
          int M, int Kpad, int N, int ostride, int relu) {
    const int lane = threadIdx.x & 31;
    const int wv   = threadIdx.x >> 5;
    const int tile = blockIdx.x * 4 + wv;
    if (tile * 16 >= N) return;
    const int mtile = blockIdx.y;
    const int nkc   = Kpad >> 5;
    const int nct   = M >> 4;
    const int n  = lane & 15;
    const int hi = lane >> 4;
    const int p  = tile * 16 + n;                       // point (column)
    const half_t* brow = B + (long)p * Kpad;

    v8f acc = {};
    for (int kcb = 0; kcb < nkc; ++kcb) {
        const int kb = kcb * 32 + hi * 16;
        v16h bfrag = *(const v16h*)(brow + kb);
        v16h afrag = Ap[((long)kcb * nct + mtile) * 32 + lane];
        acc = __builtin_amdgcn_wmma_f32_16x16x32_f16(false, afrag, false, bfrag,
                                                     (short)0, acc, false, false);
    }
    half_t* dst = out + (long)p * ostride;
    const int mrow0 = hi * 8;
    #pragma unroll
    for (int v = 0; v < 8; ++v) {
        const int co = mtile * 16 + mrow0 + v;
        float r = acc[v] + bias[co];
        if (relu) r = fmaxf(r, 0.f);
        dst[co] = (half_t)r;
    }
}

// ---------------------------------------------------------------------------
// Host side
// ---------------------------------------------------------------------------
extern "C" void kernel_launch(void* const* d_in, const int* in_sizes, int n_in,
                              void* d_out, int out_size, void* d_ws, size_t ws_size,
                              hipStream_t stream) {
    (void)in_sizes; (void)n_in; (void)out_size; (void)ws_size;
    const int*   coord  = (const int*)d_in[0];
    const int*   iidx   = (const int*)d_in[1];
    const float* tgrid  = (const float*)d_in[2];
    const float* pgrid  = (const float*)d_in[3];
    const float* vdir   = (const float*)d_in[4];
    const float* lpts   = (const float*)d_in[5];
    const float* cemb   = (const float*)d_in[6];
    const float* wc0    = (const float*)d_in[7];
    const float* wd0    = (const float*)d_in[8];
    const float* wc1    = (const float*)d_in[9];
    const float* wd1    = (const float*)d_in[10];
    const float* wc2    = (const float*)d_in[11];
    const float* bn2s   = (const float*)d_in[12];
    const float* bn2b   = (const float*)d_in[13];
    const float* wd2    = (const float*)d_in[14];
    const float* wc3    = (const float*)d_in[15];
    const float* bn3s   = (const float*)d_in[16];
    const float* bn3b   = (const float*)d_in[17];
    const float* wd3    = (const float*)d_in[18];
    const float* wc4    = (const float*)d_in[19];
    const float* bn4s   = (const float*)d_in[20];
    const float* bn4b   = (const float*)d_in[21];
    const float* lemb   = (const float*)d_in[22];
    const float* fc0w   = (const float*)d_in[23];
    const float* fc0b   = (const float*)d_in[24];
    const float* fc1w   = (const float*)d_in[25];
    const float* fc1b   = (const float*)d_in[26];
    const float* fc2w   = (const float*)d_in[27];
    const float* fc2b   = (const float*)d_in[28];
    const float* alw    = (const float*)d_in[29];
    const float* alb    = (const float*)d_in[30];
    const float* few    = (const float*)d_in[31];
    const float* feb    = (const float*)d_in[32];
    const float* lfw    = (const float*)d_in[33];
    const float* lfb    = (const float*)d_in[34];
    const float* viw    = (const float*)d_in[35];
    const float* vib    = (const float*)d_in[36];
    const float* rgw    = (const float*)d_in[37];
    const float* rgb    = (const float*)d_in[38];
    float* out = (float*)d_out;

    const int Np = 16384, Nt = 110592, Nv = 6890;

    // byte-based workspace allocator, 64B-aligned regions
    char* wsb = (char*)d_ws;
    size_t off = 0;
    auto alloc = [&](size_t bytes) {
        void* p = wsb + off;
        off += (bytes + 63) & ~(size_t)63;
        return p;
    };
    half_t* A0  = (half_t*)alloc(442368L * 16 * 2);   // level0 ping (f16)
    half_t* A1  = (half_t*)alloc(442368L * 16 * 2);
    float*  M0  = (float*)alloc(442368L * 4);
    half_t* B0  = (half_t*)alloc(55296L * 32 * 2);
    half_t* B1  = (half_t*)alloc(55296L * 32 * 2);
    float*  M1  = (float*)alloc(55296L * 4);
    half_t* C0  = (half_t*)alloc(6912L * 64 * 2);
    half_t* C1  = (half_t*)alloc(6912L * 64 * 2);
    float*  M2  = (float*)alloc(6912L * 4);
    half_t* Dl0 = (half_t*)alloc(864L * 128 * 2);
    half_t* Dl1 = (half_t*)alloc(864L * 128 * 2);
    float*  M3  = (float*)alloc(864L * 4);
    half_t* E0  = (half_t*)alloc(108L * 128 * 2);
    half_t* E1  = (half_t*)alloc(108L * 128 * 2);
    float*  M4  = (float*)alloc(108L * 4);
    half_t* TV  = (half_t*)alloc((long)Nt * 128 * 2); // tgrid-sampled volume (reused)
    half_t* FE  = (half_t*)alloc((long)Np * 352 * 2); // concat features
    half_t* H0  = (half_t*)alloc((long)Np * 256 * 2);
    half_t* H1  = (half_t*)alloc((long)Np * 256 * 2);
    half_t* H2  = (half_t*)alloc((long)Np * 256 * 2);
    half_t* FT  = (half_t*)alloc((long)Np * 384 * 2); // features ++ latent
    half_t* V2  = (half_t*)alloc((long)Np * 352 * 2); // latfc ++ view ++ light (pad->352)
    half_t* H3  = (half_t*)alloc((long)Np * 128 * 2);
    half_t* ZB  = (half_t*)alloc(64);                 // 64B zero page for OOB taps

    // packed weight fragments (f16)
    auto pbytes = [](int ntaps, int Ci, int Co) {
        return (size_t)ntaps * ((Ci + 31) / 32) * (Co / 16) * 512 * 2;
    };
    half_t* Pc0[2]; for (int i = 0; i < 2; ++i) Pc0[i] = (half_t*)alloc(pbytes(27, 16, 16));
    half_t* Pc1[2]; for (int i = 0; i < 2; ++i) Pc1[i] = (half_t*)alloc(pbytes(27, 32, 32));
    half_t* Pc2[3]; for (int i = 0; i < 3; ++i) Pc2[i] = (half_t*)alloc(pbytes(27, 64, 64));
    half_t* Pc3[3]; for (int i = 0; i < 3; ++i) Pc3[i] = (half_t*)alloc(pbytes(27, 128, 128));
    half_t* Pc4[3]; for (int i = 0; i < 3; ++i) Pc4[i] = (half_t*)alloc(pbytes(27, 128, 128));
    half_t* Pd0 = (half_t*)alloc(pbytes(8, 16, 32));
    half_t* Pd1 = (half_t*)alloc(pbytes(8, 32, 64));
    half_t* Pd2 = (half_t*)alloc(pbytes(8, 64, 128));
    half_t* Pd3 = (half_t*)alloc(pbytes(8, 128, 128));
    half_t* Pf0 = (half_t*)alloc(pbytes(1, 352, 256));
    half_t* Pf1 = (half_t*)alloc(pbytes(1, 256, 256));
    half_t* Pf2 = (half_t*)alloc(pbytes(1, 256, 256));
    half_t* Pfe = (half_t*)alloc(pbytes(1, 256, 256));
    half_t* Plf = (half_t*)alloc(pbytes(1, 384, 256));
    half_t* Pvw = (half_t*)alloc(pbytes(1, 346, 128));

    auto z = [&](void* p, long nfloats) {
        zero_kernel<<<(unsigned)((nfloats + 255) / 256), 256, 0, stream>>>((float*)p, nfloats);
    };
    z(A0, 442368L * 16 / 2);
    z(M0, 442368);
    z(V2, (long)Np * 352 / 2);  // zero K-padding for the view GEMM
    z(ZB, 16);

    auto packw = [&](const float* src, half_t* dst, int ntaps, int Ci, int Co, int tr) {
        int total = ntaps * ((Ci + 31) / 32) * (Co / 16) * 32;
        packw_kernel<<<(total + 127) / 128, 128, 0, stream>>>(src, dst, ntaps, Ci, Co, tr);
    };
    packw(wc0,                       Pc0[0], 27, 16, 16, 0);
    packw(wc0 + 27 * 16 * 16,        Pc0[1], 27, 16, 16, 0);
    packw(wc1,                       Pc1[0], 27, 32, 32, 0);
    packw(wc1 + 27 * 32 * 32,        Pc1[1], 27, 32, 32, 0);
    for (int i = 0; i < 3; ++i) packw(wc2 + (long)i * 27 * 64 * 64,   Pc2[i], 27, 64, 64, 0);
    for (int i = 0; i < 3; ++i) packw(wc3 + (long)i * 27 * 128 * 128, Pc3[i], 27, 128, 128, 0);
    for (int i = 0; i < 3; ++i) packw(wc4 + (long)i * 27 * 128 * 128, Pc4[i], 27, 128, 128, 0);
    packw(wd0, Pd0, 8, 16, 32, 0);
    packw(wd1, Pd1, 8, 32, 64, 0);
    packw(wd2, Pd2, 8, 64, 128, 0);
    packw(wd3, Pd3, 8, 128, 128, 0);
    packw(fc0w, Pf0, 1, 352, 256, 1);
    packw(fc1w, Pf1, 1, 256, 256, 1);
    packw(fc2w, Pf2, 1, 256, 256, 1);
    packw(few,  Pfe, 1, 256, 256, 1);
    packw(lfw,  Plf, 1, 384, 256, 1);
    packw(viw,  Pvw, 1, 346, 128, 1);

    scatter_kernel<<<(Nv + 127) / 128, 128, 0, stream>>>(coord, cemb, A0, M0, Nv, 96, 96);

    auto conv = [&](const half_t* in, const half_t* wp, const float* mk,
                    const float* sc, const float* sh, half_t* o,
                    int d, int h, int wdim, int ci, int co) {
        int tiles = (d * h * wdim + 15) / 16;
        dim3 g((tiles + 3) / 4, co / 16);
        conv3x3_wmma<<<g, 128, 0, stream>>>(in, (const v16h*)wp, mk, sc, sh, ZB, o,
                                            d, h, wdim, ci, co);
    };
    auto down = [&](const half_t* in, const half_t* wp, half_t* o,
                    int d, int h, int wdim, int ci, int co) {
        int tiles = (d * h * wdim + 15) / 16;
        dim3 g((tiles + 3) / 4, co / 16);
        convdown_wmma<<<g, 128, 0, stream>>>(in, (const v16h*)wp, ZB, o, d, h, wdim, ci, co);
    };

    // level 0: double conv (16ch, 48x96x96)
    conv(A0, Pc0[0], M0, nullptr, nullptr, A1, 48, 96, 96, 16, 16);
    conv(A1, Pc0[1], M0, nullptr, nullptr, A0, 48, 96, 96, 16, 16);
    down(A0, Pd0, B0, 24, 48, 48, 16, 32);
    maskpool_kernel<<<(55296 + 255) / 256, 256, 0, stream>>>(M0, M1, 24, 48, 48);

    // level 1: double conv (32ch, 24x48x48) -> f1
    conv(B0, Pc1[0], M1, nullptr, nullptr, B1, 24, 48, 48, 32, 32);
    conv(B1, Pc1[1], M1, nullptr, nullptr, B0, 24, 48, 48, 32, 32);
    gsample_kernel<<<Nt, 64, 0, stream>>>(B0, 24, 48, 48, 32, tgrid, Nt, TV, 32, 0);
    gsample_kernel<<<Np, 64, 0, stream>>>(TV, 48, 48, 48, 32, pgrid, Np, FE, 352, 0);
    down(B0, Pd1, C0, 12, 24, 24, 32, 64);
    maskpool_kernel<<<(6912 + 255) / 256, 256, 0, stream>>>(M1, M2, 12, 24, 24);

    // level 2: triple conv + bn (64ch, 12x24x24) -> f2
    conv(C0, Pc2[0], M2, bn2s,       bn2b,       C1, 12, 24, 24, 64, 64);
    conv(C1, Pc2[1], M2, bn2s + 64,  bn2b + 64,  C0, 12, 24, 24, 64, 64);
    conv(C0, Pc2[2], M2, bn2s + 128, bn2b + 128, C1, 12, 24, 24, 64, 64);
    gsample_kernel<<<Nt, 64, 0, stream>>>(C1, 12, 24, 24, 64, tgrid, Nt, TV, 64, 0);
    gsample_kernel<<<Np, 64, 0, stream>>>(TV, 48, 48, 48, 64, pgrid, Np, FE, 352, 32);
    down(C1, Pd2, Dl0, 6, 12, 12, 64, 128);
    maskpool_kernel<<<(864 + 255) / 256, 256, 0, stream>>>(M2, M3, 6, 12, 12);

    // level 3: triple conv + bn (128ch, 6x12x12) -> f3
    conv(Dl0, Pc3[0], M3, bn3s,       bn3b,       Dl1, 6, 12, 12, 128, 128);
    conv(Dl1, Pc3[1], M3, bn3s + 128, bn3b + 128, Dl0, 6, 12, 12, 128, 128);
    conv(Dl0, Pc3[2], M3, bn3s + 256, bn3b + 256, Dl1, 6, 12, 12, 128, 128);
    gsample_kernel<<<Nt, 64, 0, stream>>>(Dl1, 6, 12, 12, 128, tgrid, Nt, TV, 128, 0);
    gsample_kernel<<<Np, 64, 0, stream>>>(TV, 48, 48, 48, 128, pgrid, Np, FE, 352, 96);
    down(Dl1, Pd3, E0, 3, 6, 6, 128, 128);
    maskpool_kernel<<<1, 128, 0, stream>>>(M3, M4, 3, 6, 6);

    // level 4: triple conv + bn (128ch, 3x6x6) -> f4
    conv(E0, Pc4[0], M4, bn4s,       bn4b,       E1, 3, 6, 6, 128, 128);
    conv(E1, Pc4[1], M4, bn4s + 128, bn4b + 128, E0, 3, 6, 6, 128, 128);
    conv(E0, Pc4[2], M4, bn4s + 256, bn4b + 256, E1, 3, 6, 6, 128, 128);
    gsample_kernel<<<Nt, 64, 0, stream>>>(E1, 3, 6, 6, 128, tgrid, Nt, TV, 128, 0);
    gsample_kernel<<<Np, 64, 0, stream>>>(TV, 48, 48, 48, 128, pgrid, Np, FE, 352, 224);

    // per-point MLP
    auto gem = [&](const half_t* Ap, const half_t* Bm, const float* bi, half_t* o,
                   int M, int Kp, int ost, int rl) {
        dim3 g((Np / 16 + 3) / 4, M / 16);
        gemm_wmma<<<g, 128, 0, stream>>>((const v16h*)Ap, Bm, bi, o, M, Kp, Np, ost, rl);
    };
    gem(Pf0, FE, fc0b, H0, 256, 352, 256, 1);
    gem(Pf1, H0, fc1b, H1, 256, 256, 256, 1);
    gem(Pf2, H1, fc2b, H2, 256, 256, 256, 1);
    alpha_kernel<<<(Np + 127) / 128, 128, 0, stream>>>(H2, alw, alb, out, Np);
    gem(Pfe, H2, feb, FT, 256, 256, 384, 0);               // features -> FT[:,0:256]
    latfill_kernel<<<Np, 128, 0, stream>>>(lemb, iidx, FT, 384, 256, 128);
    gem(Plf, FT, lfb, V2, 256, 384, 352, 0);               // latfc -> V2[:,0:256]
    catvl_kernel<<<Np, 96, 0, stream>>>(vdir, lpts, V2);   // V2[:,256:346]
    gem(Pvw, V2, vib, H3, 128, 352, 128, 1);               // view layer (K padded 346->352)
    rgb_kernel<<<(Np + 127) / 128, 128, 0, stream>>>(H3, rgw, rgb, out, Np);
}
